// NTXentLoss_16020228014904
// MI455X (gfx1250) — matrix operand order repaired and checked by reading, not compiled, not run on hardware
//
#include <hip/hip_runtime.h>
#include <hip/hip_bf16.h>

// ---------------------------------------------------------------------------
// NT-Xent loss, fused flash-LSE formulation for MI455X (gfx1250, wave32).
//   sim = (z z^T) / T,  diag masked, loss = mean_k( logsumexp(sim[k]) - sim[k, k^B] )
// Phase 1: fp32 -> bf16 conversion of z = [z_i; z_j]  (4 MB, L2-resident)
// Phase 2: per 16-row tile, stream 1024 column tiles through
//          v_wmma_f32_16x16x32_bf16 (K=128 = 4 chained WMMAs) with online LSE.
//   Hot-loop LSE: 1 raw v_exp_f32 per element (no OCML subnormal expansion);
//   diag/pos handling on two SALU-uniform special tiles; max-records in a
//   rare EXEC-masked branch.
// ---------------------------------------------------------------------------

typedef __attribute__((ext_vector_type(16))) __bf16 v16bf;
typedef __attribute__((ext_vector_type(8)))  __bf16 v8bf;
typedef __attribute__((ext_vector_type(8)))  float  v8f;

#define NTX_B      8192
#define NTX_D      128
#define NTX_N      16384            // 2*B
#define NTX_NT     1024             // N / 16 column tiles
#define NTX_SCALE  2.8853900817779268f   // (1/T) * log2(e), T = 0.5
#define NTX_LN2    0.6931471805599453f

// Raw v_exp_f32 / v_log_f32 (base-2, flush-to-zero): subnormal exp2 results
// are irrelevant to a logsumexp, so skip the OCML range expansion.
__device__ __forceinline__ float fast_exp2(float x) { return __builtin_amdgcn_exp2f(x); }
__device__ __forceinline__ float fast_log2(float x) { return __builtin_amdgcn_logf(x); }

__global__ void ntxent_zero_out(float* __restrict__ out) {
    out[0] = 0.0f;
}

// Convert z_i and z_j (fp32) into one contiguous [N, D] bf16 matrix (RNE).
__global__ void ntxent_cvt_bf16(const float* __restrict__ zi,
                                const float* __restrict__ zj,
                                unsigned int* __restrict__ outw) {
    const int t = blockIdx.x * blockDim.x + threadIdx.x;      // one float4 per thread
    const long base = (long)t * 4;
    const long half = (long)NTX_B * NTX_D;                    // 1048576
    const float4 f = (base < half) ? ((const float4*)zi)[t]
                                   : ((const float4*)(zj))[t - (half >> 2)];
    unsigned int u0 = __float_as_uint(f.x), u1 = __float_as_uint(f.y);
    unsigned int u2 = __float_as_uint(f.z), u3 = __float_as_uint(f.w);
    u0 = (u0 + 0x7FFFu + ((u0 >> 16) & 1u)) >> 16;
    u1 = (u1 + 0x7FFFu + ((u1 >> 16) & 1u)) >> 16;
    u2 = (u2 + 0x7FFFu + ((u2 >> 16) & 1u)) >> 16;
    u3 = (u3 + 0x7FFFu + ((u3 >> 16) & 1u)) >> 16;
    uint2 p;
    p.x = u0 | (u1 << 16);
    p.y = u2 | (u3 << 16);
    ((uint2*)outw)[t] = p;
}

// 4 waves per block; each wave owns one 16-row tile of sim.
__global__ void __launch_bounds__(128)
ntxent_main(const __bf16* __restrict__ zb, float* __restrict__ out) {
    const int lane  = threadIdx.x & 31;
    const int wave  = threadIdx.x >> 5;
    // Wave-uniform row-tile index in an SGPR so the special-tile test is SALU.
    const int mtile = __builtin_amdgcn_readfirstlane(blockIdx.x * 4 + wave); // 0..1023
    const int l15   = lane & 15;
    const int khalf = lane >> 4;

    // ---- Preload A fragments: rows mtile*16 + l15, all of K=128 -------------
    // k order used here (contiguous, same as B): lanes 0-15 hold K run [0..15], lanes 16-31 [16..31]
    const __bf16* arow = zb + (size_t)(mtile * 16 + l15) * NTX_D;
    v16bf afrag[4];
#pragma unroll
    for (int c = 0; c < 4; ++c) {
        const __bf16* p = arow + c * 32;
        v8bf lo = *(const v8bf*)(p + khalf * 16);         // K c*32 + [0..7]/[16..23]
        v8bf hi = *(const v8bf*)(p + 8 + khalf * 16);     // K c*32 + [8..15]/[24..31]
        afrag[c] = __builtin_shufflevector(lo, hi, 0,1,2,3,4,5,6,7,8,9,10,11,12,13,14,15);
    }

    // Online LSE state per (lane, j): row = mtile*16 + j + 8*khalf,
    // columns seen = l15 + 16*ntile.  nm = -runningMax (log2 domain).
    float nm[8], lrun[8], pos[8];
#pragma unroll
    for (int j = 0; j < 8; ++j) { nm[j] = 1.0e4f; lrun[j] = 0.0f; pos[j] = 0.0f; }
    const int mrowbase = mtile * 16 + 8 * khalf;
    const int postile  = mtile ^ (NTX_B / 16);            // tile holding positive col

    for (int ntile = 0; ntile < NTX_NT; ++ntile) {
        // Prefetch B rows 8 tiles ahead (row = 256 B -> two 128 B lines).
        {
            const int pft = (ntile + 8 < NTX_NT) ? (ntile + 8) : (NTX_NT - 1);
            const __bf16* pf = zb + (size_t)(pft * 16 + l15) * NTX_D;
            __builtin_prefetch(pf, 0, 1);
            __builtin_prefetch(pf + 64, 0, 1);
        }

        // ---- B fragments: column n of B = row n of z (k order used here, same as A:
        //      lane = N, lanes 0-15 hold K run [0..15], lanes 16-31 [16..31]) ----
        const __bf16* brow = zb + (size_t)(ntile * 16 + l15) * NTX_D;
        v16bf bfrag[4];
#pragma unroll
        for (int c = 0; c < 4; ++c) {
            const __bf16* p = brow + c * 32 + khalf * 16;
            v8bf lo = *(const v8bf*)(p);
            v8bf hi = *(const v8bf*)(p + 8);
            bfrag[c] = __builtin_shufflevector(lo, hi, 0,1,2,3,4,5,6,7,8,9,10,11,12,13,14,15);
        }

        v8f acc = {0.f, 0.f, 0.f, 0.f, 0.f, 0.f, 0.f, 0.f};
#pragma unroll
        for (int c = 0; c < 4; ++c) {
            acc = __builtin_amdgcn_wmma_f32_16x16x32_bf16(
                /*neg_a=*/false, afrag[c], /*neg_b=*/false, bfrag[c],
                /*c_mod=*/(short)0, acc, /*reuse_a=*/false, /*reuse_b=*/false);
        }

        if (ntile == mtile || ntile == postile) {
            // SALU-uniform slow path: diagonal mask + positive-pair capture (2 tiles).
            const int col = ntile * 16 + l15;
#pragma unroll
            for (int j = 0; j < 8; ++j) {
                const int row = mrowbase + j;
                float s = acc[j] * NTX_SCALE;
                s = (col == row) ? -3.0e38f : s;              // diag -> exp2 == 0
                pos[j] += (col == (row ^ NTX_B)) ? s : 0.0f;  // positive pair col
                const float m  = -nm[j];
                const float mx = fmaxf(m, s);
                lrun[j] = lrun[j] * fast_exp2(m - mx) + fast_exp2(s - mx);
                nm[j]   = -mx;
            }
        } else {
            // Hot path: 1 raw v_exp_f32 per element; rare rescale branch on new max.
            float d[8];
#pragma unroll
            for (int j = 0; j < 8; ++j)
                d[j] = __builtin_fmaf(acc[j], NTX_SCALE, nm[j]);   // s - m
#pragma unroll
            for (int j = 0; j < 8; ++j)
                lrun[j] += fast_exp2(fminf(d[j], 0.0f));   // d>0 adds the "+1" term
            float dmax = fmaxf(fmaxf(fmaxf(d[0], d[1]), fmaxf(d[2], d[3])),
                               fmaxf(fmaxf(d[4], d[5]), fmaxf(d[6], d[7])));
            if (dmax > 0.0f) {                         // EXEC-masked, rare
#pragma unroll
                for (int j = 0; j < 8; ++j) {
                    const bool up = d[j] > 0.0f;
                    const float r = __builtin_fmaf(lrun[j] - 1.0f, fast_exp2(-d[j]), 1.0f);
                    lrun[j] = up ? r : lrun[j];
                    nm[j]   = up ? (nm[j] - d[j]) : nm[j];
                }
            }
        }
    }

    // ---- Merge (m, l, pos) across the 16 lanes of each half, sum rows -------
    float wsum = 0.0f;
#pragma unroll
    for (int j = 0; j < 8; ++j) {
        float mj = -nm[j], lj = lrun[j], pj = pos[j];
#pragma unroll
        for (int off = 8; off >= 1; off >>= 1) {          // stays within 16-lane half
            const float om = __shfl_xor(mj, off, 32);
            const float ol = __shfl_xor(lj, off, 32);
            const float op = __shfl_xor(pj, off, 32);
            const float mn = fmaxf(mj, om);
            lj = lj * fast_exp2(mj - mn) + ol * fast_exp2(om - mn);
            mj = mn;
            pj += op;
        }
        if (l15 == 0) wsum += (mj + fast_log2(lj) - pj);  // per-row (lse - pos), base 2
    }

    __shared__ float spart[8];                            // 4 waves x 2 halves
    if (l15 == 0) spart[wave * 2 + khalf] = wsum;
    __syncthreads();
    if (threadIdx.x == 0) {
        float t = 0.0f;
#pragma unroll
        for (int i = 0; i < 8; ++i) t += spart[i];
        atomicAdd(out, t * (NTX_LN2 / (float)NTX_N));     // back to nats, /N
    }
}

extern "C" void kernel_launch(void* const* d_in, const int* in_sizes, int n_in,
                              void* d_out, int out_size, void* d_ws, size_t ws_size,
                              hipStream_t stream) {
    const float* zi = (const float*)d_in[0];
    const float* zj = (const float*)d_in[1];
    float* out = (float*)d_out;
    unsigned int* wsw = (unsigned int*)d_ws;              // bf16 z buffer: N*D*2 = 4 MB

    ntxent_zero_out<<<1, 1, 0, stream>>>(out);

    const int total_f4 = (NTX_N * NTX_D) / 4;             // 524288 float4s
    ntxent_cvt_bf16<<<total_f4 / 256, 256, 0, stream>>>(zi, zj, wsw);

    ntxent_main<<<NTX_NT / 4, 128, 0, stream>>>((const __bf16*)d_ws, out);
}
